// SpatialAttention_90864328114316
// MI455X (gfx1250) — matrix operand order repaired and checked by reading, not compiled, not run on hardware
//
#include <hip/hip_runtime.h>
#include <hip/hip_bf16.h>

typedef __attribute__((ext_vector_type(16))) __bf16        v16bf;
typedef __attribute__((ext_vector_type(8)))  float         v8f;
typedef __attribute__((ext_vector_type(4)))  unsigned int  v4u;

union Frag {
  v16bf  v;
  v4u    q[2];     // two 16-byte halves
  __bf16 h[16];
};

// Geometry
#define NBATCH 32
#define CCH    256
#define HW     4096        // 64*64
#define WIMG   64
#define HD     64
#define STRIDE 6
#define DIL    3

__device__ __forceinline__ int pos_of(int pi, int pj, int l) {
  int ki = l >> 2, kj = l & 3;
  return (STRIDE * pi + DIL * ki) * WIMG + (STRIDE * pj + DIL * kj);
}

// ---------------------------------------------------------------------------
// Prep: fuse+convert weights to bf16 table wt[1024][256]
//   rows 0..255 = w_q, 256..767 = w_kv, 768..1023 = w_proj; biasc[1024] f32.
// ---------------------------------------------------------------------------
__global__ __launch_bounds__(256) void prep_w_kernel(
    const float* __restrict__ w_q, const float* __restrict__ b_q,
    const float* __restrict__ w_kv, const float* __restrict__ b_kv,
    const float* __restrict__ w_proj, const float* __restrict__ b_proj,
    __bf16* __restrict__ wt, float* __restrict__ biasc)
{
  const int row = blockIdx.x;
  const int c   = threadIdx.x;
  const float* wsrc; const float* bsrc; int r;
  if (row < 256)      { wsrc = w_q;    bsrc = b_q;    r = row;       }
  else if (row < 768) { wsrc = w_kv;   bsrc = b_kv;   r = row - 256; }
  else                { wsrc = w_proj; bsrc = b_proj; r = row - 768; }
  wt[(size_t)row * CCH + c] = (__bf16)wsrc[(size_t)r * CCH + c];
  if (c == 0) biasc[row] = bsrc[r];
}

// ---------------------------------------------------------------------------
// Transpose+convert: q[n][c][hw] f32 -> qT[n][hw][c] bf16 (32x32 LDS tiles).
// grid (128 hw-tiles, 8 c-tiles, 32 n), block 256.
// ---------------------------------------------------------------------------
__global__ __launch_bounds__(256) void qT_kernel(
    const float* __restrict__ q, __bf16* __restrict__ qT)
{
  __shared__ __bf16 tile[32][33];
  const int n   = blockIdx.z;
  const int c0  = blockIdx.y * 32;
  const int m0  = blockIdx.x * 32;
  const int lc  = threadIdx.x >> 5;   // 0..7
  const int lm  = threadIdx.x & 31;   // 0..31

  const float* src = q + ((size_t)n * CCH + c0) * HW + m0;
#pragma unroll
  for (int i = 0; i < 4; ++i) {
    const int c = lc + i * 8;
    tile[c][lm] = (__bf16)src[(size_t)c * HW + lm];     // coalesced along hw
  }
  __syncthreads();
  __bf16* dst = qT + ((size_t)n * HW + m0) * CCH + c0;
#pragma unroll
  for (int i = 0; i < 4; ++i) {
    const int m = lc + i * 8;
    dst[(size_t)m * CCH + lm] = tile[lm][m];            // coalesced along c
  }
}

// ---------------------------------------------------------------------------
// QKV projection GEMM with 2x2 register tiling: each wave computes a
// 32(o) x 32(m) tile (4 WMMA accumulators) so every k-step issues
// 8 b128 loads for 4 WMMAs (2 loads/WMMA).  K=256 in 8 steps.
// Output staged through a 32x32 LDS tile -> qkvT[n][m][768] (4 b128/lane).
// grid (4096 m-pairs, 3), block (32, 8): 24 o-slots x 4096 m-pairs.
// ---------------------------------------------------------------------------
__global__ __launch_bounds__(256) void qkv_gemm_kernel(
    const __bf16* __restrict__ wt, const float* __restrict__ biasc,
    const __bf16* __restrict__ qT, __bf16* __restrict__ qkvT)
{
  __shared__ alignas(16) __bf16 sT[8][32][32];   // [wave][m][o]
  const int lane = threadIdx.x;
  const int wv   = threadIdx.y;
  const int half = lane >> 4;
  const int l15  = lane & 15;
  const int oslot = blockIdx.y * 8 + wv;        // 0..23  -> o in [oslot*32, +32)
  const int mt2   = blockIdx.x;                 // 0..4095 -> 32 m's
  const int n     = mt2 >> 7;
  const int hw0   = (mt2 & 127) << 5;

  const __bf16* wrow0 = wt + (size_t)(oslot * 32 + l15) * CCH;
  const __bf16* wrow1 = wrow0 + (size_t)16 * CCH;
  const __bf16* xrow0 = qT + ((size_t)n * HW + hw0 + l15) * CCH;
  const __bf16* xrow1 = xrow0 + (size_t)16 * CCH;

  v8f acc00 = {0.f,0.f,0.f,0.f,0.f,0.f,0.f,0.f};
  v8f acc01 = acc00, acc10 = acc00, acc11 = acc00;

  for (int k0 = 0; k0 < CCH; k0 += 32) {
    __builtin_prefetch(xrow0 + k0 + 64, 0, 1);          // global_prefetch_b8
    Frag a0, a1, b0, b1;
    a0.q[0] = *(const v4u*)(wrow0 + k0 + 8 * half);     // A: K=8h+e / 8h+16+e
    a0.q[1] = *(const v4u*)(wrow0 + k0 + 16 + 8 * half);
    a1.q[0] = *(const v4u*)(wrow1 + k0 + 8 * half);
    a1.q[1] = *(const v4u*)(wrow1 + k0 + 16 + 8 * half);
    b0.q[0] = *(const v4u*)(xrow0 + k0 + 8 * half);     // B: K=8h+e / 8h+16+e
    b0.q[1] = *(const v4u*)(xrow0 + k0 + 16 + 8 * half);
    b1.q[0] = *(const v4u*)(xrow1 + k0 + 8 * half);
    b1.q[1] = *(const v4u*)(xrow1 + k0 + 16 + 8 * half);
    acc00 = __builtin_amdgcn_wmma_f32_16x16x32_bf16(false, a0.v, false, b0.v,
                                                    (short)0, acc00, false, false);
    acc01 = __builtin_amdgcn_wmma_f32_16x16x32_bf16(false, a0.v, false, b1.v,
                                                    (short)0, acc01, false, false);
    acc10 = __builtin_amdgcn_wmma_f32_16x16x32_bf16(false, a1.v, false, b0.v,
                                                    (short)0, acc10, false, false);
    acc11 = __builtin_amdgcn_wmma_f32_16x16x32_bf16(false, a1.v, false, b1.v,
                                                    (short)0, acc11, false, false);
  }
  // D tile (oi,mj): M = r+8h (o-local), N = l15 (m-local)
#pragma unroll
  for (int r = 0; r < 8; ++r) {
    const int ob = oslot * 32;
    const int ol = r + 8 * half;
    sT[wv][l15     ][ol     ] = (__bf16)(acc00[r] + biasc[ob + ol]);
    sT[wv][l15 + 16][ol     ] = (__bf16)(acc01[r] + biasc[ob + ol]);
    sT[wv][l15     ][ol + 16] = (__bf16)(acc10[r] + biasc[ob + 16 + ol]);
    sT[wv][l15 + 16][ol + 16] = (__bf16)(acc11[r] + biasc[ob + 16 + ol]);
  }
  __syncthreads();
  __bf16* dst = qkvT + ((size_t)n * HW + hw0 + lane) * 768 + oslot * 32;
#pragma unroll
  for (int i = 0; i < 4; ++i)                           // 4 x b128 per lane
    *(v4u*)(dst + i * 8) = *(const v4u*)&sT[wv][lane][i * 8];
}

// ---------------------------------------------------------------------------
// Attention per (batch, head, patch): one wave per unit.
// QK^T: 2 WMMAs (b128 operand loads).  Softmax (scale 1/8) via per-wave LDS.
// attn @ V^T: 4 WMMAs, K padded 16->32 with zeros.  Fold: coalesced f32 atomics.
// grid 1600, block (32, 8): exactly one unit per wave.
// ---------------------------------------------------------------------------
__global__ __launch_bounds__(256) void attn_kernel(
    const __bf16* __restrict__ qkvT, float* __restrict__ foldedT)
{
  __shared__ float                 sS[8][16][16];
  __shared__ alignas(16) __bf16    sA[8][16][32];

  const int lane = threadIdx.x;
  const int wv   = threadIdx.y;
  const int half = lane >> 4;
  const int l15  = lane & 15;

  const int unit = blockIdx.x * 8 + wv;
  const int b    = unit / 400;
  const int rem  = unit - b * 400;
  const int head = rem / 100;
  const int s    = rem - head * 100;
  const int pi   = s / 10;
  const int pj   = s - pi * 10;

  // zero the K=16..31 pad of the attn tile
#pragma unroll
  for (int i = 0; i < 8; ++i) {
    const int idx = lane * 8 + i;
    sA[wv][idx >> 4][16 + (idx & 15)] = (__bf16)0.0f;
  }

  const int posA = pos_of(pi, pj, l15);       // lane's patch element (x or y)
  const __bf16* rowA = qkvT + ((size_t)b * HW + posA) * 768;
  const int qoff = head * HD;                 // q channels
  const int koff = CCH + head * HD;           // k channels

  // ---- S = q^T k over d=64 (scale folded into softmax) ---------------------
  v8f accS = {0.f,0.f,0.f,0.f,0.f,0.f,0.f,0.f};
  for (int kc = 0; kc < HD; kc += 32) {
    Frag a, bm;
    a.q[0]  = *(const v4u*)(rowA + qoff + kc + 8 * half);
    a.q[1]  = *(const v4u*)(rowA + qoff + kc + 16 + 8 * half);
    bm.q[0] = *(const v4u*)(rowA + koff + kc + 8 * half);
    bm.q[1] = *(const v4u*)(rowA + koff + kc + 16 + 8 * half);
    accS = __builtin_amdgcn_wmma_f32_16x16x32_bf16(false, a.v, false, bm.v,
                                                   (short)0, accS, false, false);
  }
#pragma unroll
  for (int r = 0; r < 8; ++r) sS[wv][r + 8 * half][l15] = accS[r];
  __syncthreads();

  // ---- row softmax, hd^-0.5 = 0.125 applied here ---------------------------
  if (lane < 16) {
    float row[16], m = -3.4e38f;
#pragma unroll
    for (int y = 0; y < 16; ++y) { row[y] = sS[wv][lane][y] * 0.125f; m = fmaxf(m, row[y]); }
    float sum = 0.f;
#pragma unroll
    for (int y = 0; y < 16; ++y) { row[y] = __expf(row[y] - m); sum += row[y]; }
    const float inv = 1.f / sum;
#pragma unroll
    for (int y = 0; y < 16; ++y) sA[wv][lane][y] = (__bf16)(row[y] * inv);
  }
  __syncthreads();

  // ---- O^T = attn @ v^T : 4 d-chunks ---------------------------------------
  Frag a2;
  a2.q[0] = *(const v4u*)&sA[wv][l15][16 * half];       // ds_load_b128
  a2.q[1] = *(const v4u*)&sA[wv][l15][16 * half + 8];
  const __bf16* vbase = qkvT + (size_t)b * HW * 768 + 2 * CCH + head * HD;

  for (int d0 = 0; d0 < HD; d0 += 16) {
    Frag b2;
#pragma unroll
    for (int e = 0; e < 16; ++e) {
      const int k = 16 * half + e;                      // reduction index y
      b2.h[e] = (k < 16)
        ? vbase[(size_t)pos_of(pi, pj, k) * 768 + d0 + l15]
        : (__bf16)0.0f;
    }
    v8f accO = {0.f,0.f,0.f,0.f,0.f,0.f,0.f,0.f};
    accO = __builtin_amdgcn_wmma_f32_16x16x32_bf16(false, a2.v, false, b2.v,
                                                   (short)0, accO, false, false);
    const int c = head * HD + d0 + l15;                 // lane-contiguous channel
#pragma unroll
    for (int r = 0; r < 8; ++r) {
      const int x = r + 8 * half;
      atomicAdd(&foldedT[((size_t)b * HW + pos_of(pi, pj, x)) * CCH + c], accO[r]);
    }
  }
}

// ---------------------------------------------------------------------------
// Output projection GEMM, 2x2 register tiling (32 o x 32 m per wave).
// A from bf16 table rows 768..1023 (b128 loads); B converts f32->bf16 with
// native casts from contiguous vector loads.  grid (4096, 1), block (32, 8).
// ---------------------------------------------------------------------------
__global__ __launch_bounds__(256) void proj_gemm_kernel(
    const float* __restrict__ foldedT, const __bf16* __restrict__ wt,
    const float* __restrict__ biasc, float* __restrict__ out)
{
  const int lane = threadIdx.x;
  const int half = lane >> 4;
  const int l15  = lane & 15;
  const int oslot = threadIdx.y;                // 0..7 -> o in [oslot*32, +32)
  const int mt2   = blockIdx.x;                 // 0..4095 -> 32 m's
  const int n     = mt2 >> 7;
  const int hw0   = (mt2 & 127) << 5;

  const __bf16* wrow0 = wt + (size_t)(768 + oslot * 32 + l15) * CCH;
  const __bf16* wrow1 = wrow0 + (size_t)16 * CCH;
  const float*  xrow0 = foldedT + ((size_t)n * HW + hw0 + l15) * CCH;
  const float*  xrow1 = xrow0 + (size_t)16 * CCH;

  v8f acc00 = {0.f,0.f,0.f,0.f,0.f,0.f,0.f,0.f};
  v8f acc01 = acc00, acc10 = acc00, acc11 = acc00;

  for (int k0 = 0; k0 < CCH; k0 += 32) {
    Frag a0, a1, b0, b1;
    a0.q[0] = *(const v4u*)(wrow0 + k0 + 8 * half);
    a0.q[1] = *(const v4u*)(wrow0 + k0 + 16 + 8 * half);
    a1.q[0] = *(const v4u*)(wrow1 + k0 + 8 * half);
    a1.q[1] = *(const v4u*)(wrow1 + k0 + 16 + 8 * half);
#pragma unroll
    for (int e = 0; e < 16; ++e) {               // K=8h+e / 8h+16+e, f32 -> v_cvt bf16
      b0.h[e] = (__bf16)xrow0[k0 + (e < 8 ? 8 * half + e : 16 + 8 * half + (e - 8))];
      b1.h[e] = (__bf16)xrow1[k0 + (e < 8 ? 8 * half + e : 16 + 8 * half + (e - 8))];
    }
    acc00 = __builtin_amdgcn_wmma_f32_16x16x32_bf16(false, a0.v, false, b0.v,
                                                    (short)0, acc00, false, false);
    acc01 = __builtin_amdgcn_wmma_f32_16x16x32_bf16(false, a0.v, false, b1.v,
                                                    (short)0, acc01, false, false);
    acc10 = __builtin_amdgcn_wmma_f32_16x16x32_bf16(false, a1.v, false, b0.v,
                                                    (short)0, acc10, false, false);
    acc11 = __builtin_amdgcn_wmma_f32_16x16x32_bf16(false, a1.v, false, b1.v,
                                                    (short)0, acc11, false, false);
  }
#pragma unroll
  for (int r = 0; r < 8; ++r) {
    const int ob = oslot * 32;
    const int ol = r + 8 * half;
    const size_t base = (size_t)n * CCH * HW + hw0 + l15;
    out[base + (size_t)(ob + ol)      * HW     ] = acc00[r] + biasc[768 + ob + ol];
    out[base + (size_t)(ob + ol)      * HW + 16] = acc01[r] + biasc[768 + ob + ol];
    out[base + (size_t)(ob + 16 + ol) * HW     ] = acc10[r] + biasc[768 + ob + 16 + ol];
    out[base + (size_t)(ob + 16 + ol) * HW + 16] = acc11[r] + biasc[768 + ob + 16 + ol];
  }
}

// ---------------------------------------------------------------------------
extern "C" void kernel_launch(void* const* d_in, const int* in_sizes, int n_in,
                              void* d_out, int out_size, void* d_ws, size_t ws_size,
                              hipStream_t stream) {
  (void)in_sizes; (void)n_in; (void)out_size; (void)ws_size;
  const float* q      = (const float*)d_in[0];
  const float* w_q    = (const float*)d_in[1];
  const float* b_q    = (const float*)d_in[2];
  const float* w_kv   = (const float*)d_in[3];
  const float* b_kv   = (const float*)d_in[4];
  const float* w_proj = (const float*)d_in[5];
  const float* b_proj = (const float*)d_in[6];
  float* out = (float*)d_out;

  // workspace layout
  char* ws = (char*)d_ws;
  __bf16* wt     = (__bf16*)ws;                                   //   512 KB
  float*  biasc  = (float*)(ws + (size_t)1024 * 256 * 2);         //     4 KB
  __bf16* qT     = (__bf16*)(ws + (1u << 20));                    //    64 MB
  __bf16* qkvT   = (__bf16*)(ws + (1u << 20)
                             + (size_t)NBATCH * HW * CCH * 2);    //   192 MB
  float*  foldedT= (float*) (ws + (1u << 20)
                             + (size_t)NBATCH * HW * CCH * 2
                             + (size_t)NBATCH * HW * 768 * 2);    //   128 MB

  hipMemsetAsync(foldedT, 0, (size_t)NBATCH * HW * CCH * sizeof(float), stream);

  prep_w_kernel<<<1024, 256, 0, stream>>>(w_q, b_q, w_kv, b_kv, w_proj, b_proj,
                                          wt, biasc);
  qT_kernel<<<dim3(128, 8, NBATCH), 256, 0, stream>>>(q, qT);

  dim3 blk(32, 8);
  qkv_gemm_kernel <<<dim3(4096, 3), blk, 0, stream>>>(wt, biasc, qT, qkvT);
  attn_kernel     <<<dim3(1600),    blk, 0, stream>>>(qkvT, foldedT);
  proj_gemm_kernel<<<dim3(4096),    blk, 0, stream>>>(foldedT, wt, biasc, out);
}
